// Model_30485677867059
// MI455X (gfx1250) — compile-verified, matrix operand order checked by reading, not run on hardware
//
#include <hip/hip_runtime.h>
#include <hip/hip_bf16.h>

typedef __attribute__((ext_vector_type(16))) _Float16 v16h;
typedef __attribute__((ext_vector_type(8)))  float    v8f;

#define BATCH 16
#define TLEN  8192
#define PADL  16
#define TPAD  (TLEN + PADL)       /* 8208 */
#define CT_TILES (TPAD / 16)      /* 513  */
#define OT_TILES (TLEN / 16)      /* 512  */

// K-pattern within a 32-wide K chunk for 16-bit WMMA fragments (ISA 7.12.2).
__device__ __forceinline__ int kpat(int lane, int e) {
  return ((lane >> 4) << 3) + e + ((e >> 3) << 3);
}
// Inverse: folded kk in [0,32) -> (lane group g, element e)
__device__ __forceinline__ void kinv(int kk, int& g, int& e) {
  g = (kk >> 3) & 1;
  e = (kk & 7) + ((kk >= 16) ? 8 : 0);
}

__device__ __forceinline__ float elu1(float v) {
  return v > 0.f ? v : (__expf(v) - 1.f);
}

__device__ __forceinline__ float uhash(unsigned x) {
  x ^= x >> 16; x *= 0x7feb352dU; x ^= x >> 15; x *= 0x846ca68bU; x ^= x >> 16;
  return (float)((x >> 8) + 1u) * (1.0f / 16777217.0f);  // (0,1)
}

// ---------------------------------------------------------------------------
// Weight packers: emit fragment-ordered f16 [tile][chunk][lane][e].
// ---------------------------------------------------------------------------
template <int IC, int OTILES>
__global__ void pack_conv_w(const float* __restrict__ W, int Ovalid,
                            _Float16* __restrict__ Wp) {
  constexpr int CH = (2 * IC) / 32;
  const int total = OTILES * CH * 512;
  const int idx = blockIdx.x * blockDim.x + threadIdx.x;
  if (idx >= total) return;
  const int e = idx & 15, lane = (idx >> 4) & 31;
  const int chunk = (idx >> 9) % CH, tile = idx / (CH * 512);
  const int m = tile * 16 + (lane & 15);
  const int kk = chunk * 32 + kpat(lane, e);
  const int tap = (kk >= IC) ? 1 : 0;
  const int i = kk - tap * IC;
  Wp[idx] = (_Float16)((m < Ovalid) ? W[(m * IC + i) * 2 + tap] : 0.f);
}

template <int L>
__global__ void pack_mlp_w(const float* __restrict__ W,  // (8,64,L)
                           _Float16* __restrict__ Wp) {  // [8][4][KC][32][16]
  constexpr int KC = (L + 31) / 32;
  const int total = 8 * 4 * KC * 512;
  const int idx = blockIdx.x * blockDim.x + threadIdx.x;
  if (idx >= total) return;
  const int e = idx & 15, lane = (idx >> 4) & 31;
  const int chunk = (idx >> 9) % KC;
  const int tile = (idx / (KC * 512)) % 4;
  const int j = idx / (4 * KC * 512);
  const int m = tile * 16 + (lane & 15);
  const int l = chunk * 32 + kpat(lane, e);
  Wp[idx] = (_Float16)((l < L) ? W[(j * 64 + m) * L + l] : 0.f);
}

// ---------------------------------------------------------------------------
// Dilated causal conv as GEMM (folded K = 2*IC). 256 thr = 8 waves = 8 out
// tiles; B fragments built once per block in LDS in fragment order, then each
// wave runs ds_load_b128 + global_load_b128 + v_wmma per chunk.
// ---------------------------------------------------------------------------
template <int IC, int DIL, bool SRC_F32>
__global__ void conv_wmma_kernel(const float* __restrict__ xsrc,
                                 const _Float16* __restrict__ hsrc,
                                 const _Float16* __restrict__ Wp,  // [8][CH][32][16]
                                 const float* __restrict__ bias,
                                 _Float16* __restrict__ hdst) {   // (B,128,TPAD)
  constexpr int CH = (2 * IC) / 32;
  __shared__ _Float16 Bfr[CH * 512];
  const int blk  = blockIdx.x;
  const int b    = blk / CT_TILES;
  const int c0   = (blk % CT_TILES) * 16;
  const int tid  = threadIdx.x;
  const int lane = tid & 31;
  const int wave = tid >> 5;

  for (int idx = tid; idx < CH * 512; idx += blockDim.x) {
    const int n = idx & 15;
    const int kf = idx >> 4;             // folded k
    const int chunk = kf >> 5, kk = kf & 31;
    int g, e; kinv(kk, g, e);
    const int tap = (kf >= IC) ? 1 : 0;
    const int i = kf - tap * IC;
    const int c = c0 + n - (tap ? 0 : DIL);
    float v = 0.f;
    if constexpr (SRC_F32) {
      const int t = c - PADL;
      if (t >= 0 && t < TLEN) v = xsrc[((long)b * IC + i) * TLEN + t];
    } else {
      if (c >= 0) v = (float)hsrc[((long)b * IC + i) * TPAD + c];
    }
    Bfr[((chunk * 32 + g * 16 + n) << 4) + e] = (_Float16)v;
  }
  __syncthreads();

  if constexpr (!SRC_F32) {  // warm L2 for the next column tile
    if (c0 + 16 < TPAD)
      __builtin_prefetch(&hsrc[((long)b * IC + (tid % IC)) * TPAD + c0 + 16], 0, 1);
  }

  const _Float16* wbase = Wp + (long)wave * CH * 512;
  v8f acc = {};
#pragma unroll
  for (int chunk = 0; chunk < CH; ++chunk) {
    const v16h a  = *(const v16h*)(wbase + (chunk * 32 + lane) * 16);
    const v16h bf = *(const v16h*)(Bfr   + (chunk * 32 + lane) * 16);
    acc = __builtin_amdgcn_wmma_f32_16x16x32_f16(false, a, false, bf,
                                                 (short)0, acc, false, false);
  }
  const int m0 = wave * 16, n = lane & 15, mg = (lane >> 4) * 8;
#pragma unroll
  for (int r = 0; r < 8; ++r) {
    const int mm = m0 + mg + r;
    hdst[((long)b * 128 + mm) * TPAD + c0 + n] = (_Float16)elu1(acc[r] + bias[mm]);
  }
}

// ---------------------------------------------------------------------------
// Head convs (128 -> 9, dil=8): wave0 -> h_vecs (f32), wave1 -> h_pred (f16).
// Wpq holds 2 packed tiles (p then q), rows >= 9 pre-zeroed by the packer.
// ---------------------------------------------------------------------------
__global__ void convpq_wmma_kernel(const _Float16* __restrict__ hsrc,
                                   const _Float16* __restrict__ Wpq,  // [2][8][32][16]
                                   const float* __restrict__ Pb,
                                   const float* __restrict__ Qb,
                                   float* __restrict__ hp,        // (B,16,TPAD)
                                   _Float16* __restrict__ hq) {   // (B,16,TPAD)
  constexpr int CH = 8;
  __shared__ _Float16 Bfr[CH * 512];
  const int blk  = blockIdx.x;
  const int b    = blk / CT_TILES;
  const int c0   = (blk % CT_TILES) * 16;
  const int tid  = threadIdx.x;
  const int lane = tid & 31;
  const int wave = tid >> 5;  // 0=p, 1=q

  for (int idx = tid; idx < CH * 512; idx += blockDim.x) {
    const int n = idx & 15;
    const int kf = idx >> 4;
    const int chunk = kf >> 5, kk = kf & 31;
    int g, e; kinv(kk, g, e);
    const int tap = (kf >= 128) ? 1 : 0;
    const int i = kf - tap * 128;
    const int c = c0 + n - (tap ? 0 : 8);
    const float v = (c >= 0) ? (float)hsrc[((long)b * 128 + i) * TPAD + c] : 0.f;
    Bfr[((chunk * 32 + g * 16 + n) << 4) + e] = (_Float16)v;
  }
  __syncthreads();

  const _Float16* wbase = Wpq + (long)wave * CH * 512;
  v8f acc = {};
#pragma unroll
  for (int chunk = 0; chunk < CH; ++chunk) {
    const v16h a  = *(const v16h*)(wbase + (chunk * 32 + lane) * 16);
    const v16h bf = *(const v16h*)(Bfr   + (chunk * 32 + lane) * 16);
    acc = __builtin_amdgcn_wmma_f32_16x16x32_f16(false, a, false, bf,
                                                 (short)0, acc, false, false);
  }
  const float* bsel = wave ? Qb : Pb;
  const int n = lane & 15, mg = (lane >> 4) * 8;
#pragma unroll
  for (int r = 0; r < 8; ++r) {
    const int mm = mg + r;
    const float bi = (mm < 9) ? bsel[mm] : 0.f;
    const float v = elu1(acc[r] + bi);
    if (wave == 0) hp[((long)b * 16 + mm) * TPAD + c0 + n] = v;
    else           hq[((long)b * 16 + mm) * TPAD + c0 + n] = (_Float16)v;
  }
}

// ---------------------------------------------------------------------------
// Sticky-regime recursion + gumbel softmax (serial over T, 128-way parallel).
// ---------------------------------------------------------------------------
__global__ void scan_kernel(const float* __restrict__ hp, float* __restrict__ z) {
  const int lane = threadIdx.x;
  const int j = lane & 7;
  const int b = blockIdx.x * 4 + (lane >> 3);
  const float* base = hp + (long)b * 16 * TPAD + PADL;
  float prev = 0.f;
  for (int t = 0; t < TLEN; ++t) {
    float s = base[(long)j * TPAD + t] + 1e-6f;
    float mx = s;
#pragma unroll
    for (int o = 1; o < 8; o <<= 1) mx = fmaxf(mx, __shfl_xor(mx, o, 32));
    float ex = __expf(s - mx), sm = ex;
#pragma unroll
    for (int o = 1; o < 8; o <<= 1) sm += __shfl_xor(sm, o, 32);
    const float lp = s - mx - __logf(sm);

    float it = 0.5f + 0.5f * tanhf(base[8 * TPAD + t]);
    it = fminf(fmaxf(it, 0.01f), 0.99f);
    const float nl = (t == 0) ? lp
        : __logf(1.f - it) + lp + __logf(1.f + it * __expf(prev) / (1.f - it));
    prev = nl;

    const float u = uhash((unsigned)(((b * 8 + j) << 14) ^ t ^ 0x9e3779b9u));
    const float g = -__logf(-__logf(u));
    float y = (nl + g) * 5.0f;  // / GEN_TEMP
    float ym = y;
#pragma unroll
    for (int o = 1; o < 8; o <<= 1) ym = fmaxf(ym, __shfl_xor(ym, o, 32));
    float ey = __expf(y - ym), es = ey;
#pragma unroll
    for (int o = 1; o < 8; o <<= 1) es += __shfl_xor(es, o, 32);
    z[((long)b * 8 + j) * TLEN + t] = ey / es;
  }
}

// ---------------------------------------------------------------------------
// Regime-mixed MLP layer: y = sum_j z_j * (W_j x + B_j). X fragments built in
// LDS once per block, kept in regs across the 8 regimes; A from packed global.
// ---------------------------------------------------------------------------
template <int L, bool SRC_HQ, bool FINAL>
__global__ void mlp_wmma_kernel(const _Float16* __restrict__ X,
                                const _Float16* __restrict__ Wp,  // [8][4][KC][32][16]
                                const float* __restrict__ Bias,   // (8,64)
                                const float* __restrict__ Z,      // (B,8,T)
                                _Float16* __restrict__ Yh,
                                float* __restrict__ Ymus,
                                float* __restrict__ Ylv) {
  constexpr int KC = (L + 31) / 32;
  __shared__ _Float16 Xfr[KC * 512];
  const int blk  = blockIdx.x;
  const int b    = blk / OT_TILES;
  const int c0   = (blk % OT_TILES) * 16;
  const int tid  = threadIdx.x;
  const int lane = tid & 31;
  const int wave = tid >> 5;

  for (int idx = tid; idx < KC * 512; idx += blockDim.x) {
    const int n = idx & 15;
    const int kf = idx >> 4;
    const int chunk = kf >> 5, kk = kf & 31;
    int g, e; kinv(kk, g, e);
    float v = 0.f;
    if (kf < L) {
      if constexpr (SRC_HQ) v = (float)X[((long)b * 16 + kf) * TPAD + PADL + c0 + n];
      else                  v = (float)X[((long)b * 64 + kf) * TLEN + c0 + n];
    }
    Xfr[((chunk * 32 + g * 16 + n) << 4) + e] = (_Float16)v;
  }
  __syncthreads();

  v16h xf[KC];
#pragma unroll
  for (int q = 0; q < KC; ++q)
    xf[q] = *(const v16h*)(Xfr + (q * 32 + lane) * 16);

  const int m0 = wave * 16, n = lane & 15, g8 = (lane >> 4) * 8;
  float accf[8] = {};
  for (int j = 0; j < 8; ++j) {
    const _Float16* wj = Wp + ((long)(j * 4 + wave) * KC) * 512;
    v8f acc = {};
#pragma unroll
    for (int q = 0; q < KC; ++q) {
      const v16h a = *(const v16h*)(wj + (q * 32 + lane) * 16);
      acc = __builtin_amdgcn_wmma_f32_16x16x32_f16(false, a, false, xf[q],
                                                   (short)0, acc, false, false);
    }
    const float zj = Z[((long)b * 8 + j) * TLEN + c0 + n];
#pragma unroll
    for (int r = 0; r < 8; ++r)
      accf[r] += zj * (acc[r] + Bias[j * 64 + (m0 + g8 + r)]);
  }

#pragma unroll
  for (int r = 0; r < 8; ++r) {
    const int mm = m0 + g8 + r;
    float v = accf[r];
    if constexpr (FINAL) {
      if (mm < 32) Ymus[((long)b * 32 + mm) * TLEN + c0 + n] = v;
      else Ylv[((long)b * 32 + (mm - 32)) * TLEN + c0 + n] = fminf(fmaxf(v, -3.f), 1.f);
    } else {
      Yh[((long)b * 64 + mm) * TLEN + c0 + n] = (_Float16)v;
    }
  }
}

// ---------------------------------------------------------------------------
extern "C" void kernel_launch(void* const* d_in, const int* in_sizes, int n_in,
                              void* d_out, int out_size, void* d_ws, size_t ws_size,
                              hipStream_t stream) {
  (void)in_sizes; (void)n_in; (void)out_size; (void)ws_size;
  const float* x   = (const float*)d_in[0];
  const float* cw0 = (const float*)d_in[1];  const float* cb0 = (const float*)d_in[2];
  const float* cw1 = (const float*)d_in[3];  const float* cb1 = (const float*)d_in[4];
  const float* cw2 = (const float*)d_in[5];  const float* cb2 = (const float*)d_in[6];
  const float* pw  = (const float*)d_in[7];  const float* pb  = (const float*)d_in[8];
  const float* qw  = (const float*)d_in[9];  const float* qb  = (const float*)d_in[10];
  const float* W0  = (const float*)d_in[11]; const float* B0  = (const float*)d_in[12];
  const float* W1  = (const float*)d_in[13]; const float* B1  = (const float*)d_in[14];
  const float* W2  = (const float*)d_in[15]; const float* B2  = (const float*)d_in[16];

  float* mus = (float*)d_out;
  float* lv  = mus + (size_t)BATCH * 32 * TLEN;
  float* z   = lv  + (size_t)BATCH * 32 * TLEN;

  char* ws = (char*)d_ws;
  size_t off = 0;
  auto carve = [&](size_t bytes) { char* p = ws + off; off = (off + bytes + 255) & ~(size_t)255; return p; };
  _Float16* hA  = (_Float16*)carve((size_t)BATCH * 128 * TPAD * sizeof(_Float16));
  _Float16* hB  = (_Float16*)carve((size_t)BATCH * 128 * TPAD * sizeof(_Float16));
  float*    hp  = (float*)   carve((size_t)BATCH * 16  * TPAD * sizeof(float));
  _Float16* hq  = (_Float16*)carve((size_t)BATCH * 16  * TPAD * sizeof(_Float16));
  _Float16* p0  = (_Float16*)carve((size_t)BATCH * 64  * TLEN * sizeof(_Float16));
  _Float16* p1  = (_Float16*)carve((size_t)BATCH * 64  * TLEN * sizeof(_Float16));
  _Float16* Wc0 = (_Float16*)carve((size_t)8 * 2 * 512 * sizeof(_Float16));
  _Float16* Wc1 = (_Float16*)carve((size_t)8 * 8 * 512 * sizeof(_Float16));
  _Float16* Wc2 = (_Float16*)carve((size_t)8 * 8 * 512 * sizeof(_Float16));
  _Float16* Wpq = (_Float16*)carve((size_t)2 * 8 * 512 * sizeof(_Float16));
  _Float16* Wm0 = (_Float16*)carve((size_t)8 * 4 * 1 * 512 * sizeof(_Float16));
  _Float16* Wm1 = (_Float16*)carve((size_t)8 * 4 * 2 * 512 * sizeof(_Float16));
  _Float16* Wm2 = (_Float16*)carve((size_t)8 * 4 * 2 * 512 * sizeof(_Float16));

  // ---- pack weights into WMMA fragment order (cheap, L2-resident) ----
  pack_conv_w<32, 8><<<(8 * 2 * 512 + 255) / 256, 256, 0, stream>>>(cw0, 128, Wc0);
  pack_conv_w<128, 8><<<(8 * 8 * 512 + 255) / 256, 256, 0, stream>>>(cw1, 128, Wc1);
  pack_conv_w<128, 8><<<(8 * 8 * 512 + 255) / 256, 256, 0, stream>>>(cw2, 128, Wc2);
  pack_conv_w<128, 1><<<(8 * 512 + 255) / 256, 256, 0, stream>>>(pw, 9, Wpq);
  pack_conv_w<128, 1><<<(8 * 512 + 255) / 256, 256, 0, stream>>>(qw, 9, Wpq + 8 * 512);
  pack_mlp_w<9><<<(8 * 4 * 512 + 255) / 256, 256, 0, stream>>>(W0, Wm0);
  pack_mlp_w<64><<<(8 * 4 * 2 * 512 + 255) / 256, 256, 0, stream>>>(W1, Wm1);
  pack_mlp_w<64><<<(8 * 4 * 2 * 512 + 255) / 256, 256, 0, stream>>>(W2, Wm2);

  // ---- main pipeline ----
  const int cgrid = BATCH * CT_TILES;
  conv_wmma_kernel<32, 1, true ><<<cgrid, 256, 0, stream>>>(x, nullptr, Wc0, cb0, hA);
  conv_wmma_kernel<128, 2, false><<<cgrid, 256, 0, stream>>>(nullptr, hA, Wc1, cb1, hB);
  conv_wmma_kernel<128, 4, false><<<cgrid, 256, 0, stream>>>(nullptr, hB, Wc2, cb2, hA);
  convpq_wmma_kernel<<<cgrid, 64, 0, stream>>>(hA, Wpq, pb, qb, hp, hq);
  scan_kernel<<<4, 32, 0, stream>>>(hp, z);

  const int mgrid = BATCH * OT_TILES;
  mlp_wmma_kernel<9,  true,  false><<<mgrid, 128, 0, stream>>>(hq, Wm0, B0, z, p0, nullptr, nullptr);
  mlp_wmma_kernel<64, false, false><<<mgrid, 128, 0, stream>>>(p0, Wm1, B1, z, p1, nullptr, nullptr);
  mlp_wmma_kernel<64, false, true ><<<mgrid, 128, 0, stream>>>(p1, Wm2, B2, z, nullptr, mus, lv);
}
